// ModulatedDeformableConv2d_70703751627015
// MI455X (gfx1250) — compile-verified, matrix operand order checked by reading, not run on hardware
//
#include <hip/hip_runtime.h>
#include <hip/hip_bf16.h>

typedef __attribute__((ext_vector_type(16))) __bf16 v16bf;
typedef __attribute__((ext_vector_type(8)))  float  v8f;

#define HW (128 * 128)

static __device__ __forceinline__ unsigned short f2bf(float f) {
    unsigned u = __builtin_bit_cast(unsigned, f);
    u += 0x7FFFu + ((u >> 16) & 1u);          // round-to-nearest-even
    return (unsigned short)(u >> 16);
}

static __device__ __forceinline__ float fast_sigmoid(float v) {
    // v_exp_f32 + v_rcp_f32 instead of the exact-division chain
    return __builtin_amdgcn_rcpf(1.f + __expf(-v));
}

// A-fragment (16x32 bf16, ISA 7.12.2) index helper:
// lane L<16 -> M=L, lanes 16..31 -> M=L-16 with K offset +8;
// element s = 2*v + h; v<4 -> K base 0, v>=4 -> K base 16.
static __device__ __forceinline__ int a_klocal(int s, int laneHi) {
    int v = s >> 1, hh = s & 1;
    return ((v & 4) ? 16 : 0) + (laneHi ? 8 : 0) + ((v & 3) << 1) + hh;
}

// ---------------------------------------------------------------------------
// Pre-swizzle BOTH weight sets to bf16 WMMA A-fragment order.
//   wsA: main conv weights   (O=64  -> 4 o-tiles) x 18 K-chunks
//   wsB: offset+mod weights  (27->32, 2 o-tiles)  x 18 K-chunks (rows 27..31=0)
// Each lane's fragment is 16 contiguous bf16 (32 B).
// ---------------------------------------------------------------------------
__global__ __launch_bounds__(256) void dcn_wprep(const float* __restrict__ weight,
                                                 const float* __restrict__ offset_w,
                                                 const float* __restrict__ mod_w,
                                                 unsigned short* __restrict__ ws) {
    int tid = blockIdx.x * 256 + threadIdx.x;
    const int NA = 4 * 18 * 32 * 16;                   // 36864
    const int NB = 2 * 18 * 32 * 16;                   // 18432
    if (tid >= NA + NB) return;

    int e = (tid < NA) ? tid : (tid - NA);
    int s      = e & 15;
    int lane   = (e >> 4) & 31;
    int t_k    = (e >> 9) % 18;
    int t_o    = e / (512 * 18);
    int laneHi = lane >> 4;
    int m      = lane & 15;
    int kg     = t_k * 32 + a_klocal(s, laneHi);       // 0..575 = c*9 + k2
    int c      = kg / 9, k2 = kg % 9;

    float val;
    if (tid < NA) {
        int o = t_o * 16 + m;
        val = weight[(o * 64 + c) * 9 + k2];
    } else {
        int oc = t_o * 16 + m;                         // 0..31
        val = (oc < 18) ? offset_w[(oc * 64 + c) * 9 + k2]
            : (oc < 27) ? mod_w[((oc - 18) * 64 + c) * 9 + k2]
            : 0.f;
    }
    ws[tid] = f2bf(val);
}

// ---------------------------------------------------------------------------
// Fused deformable conv: block = (b, h, 32 consecutive w), 256 threads.
//   0a: regular-tap im2col -> Sfrag (bf16 B-fragments)
//   0b: waves 0..3: 18x WMMA -> offsets + sigmoid(mod) into LDS
//   1a: bilinear corner indices/weights per (tap,pixel)
//   1b: deformed gather+lerp -> Sfrag (reused)
//   2 : 8 waves x 18 WMMA -> modulated output
// ---------------------------------------------------------------------------
__global__ __launch_bounds__(256) void dcn_main(const float* __restrict__ x,
                                                const float* __restrict__ bias,
                                                const float* __restrict__ offset_b,
                                                const float* __restrict__ mod_b,
                                                const unsigned short* __restrict__ wsA,
                                                const unsigned short* __restrict__ wsB,
                                                float* __restrict__ out) {
    __shared__ unsigned short Sfrag[2 * 18 * 32 * 16]; // B fragments, 36864 B
    __shared__ int   cIdx[288][4];
    __shared__ float cW[288][4];
    __shared__ float offYL[9][32];
    __shared__ float offXL[9][32];
    __shared__ float modSum[32];

    const int tid = threadIdx.x;
    const int wq  = blockIdx.x & 3;
    const int h   = (blockIdx.x >> 2) & 127;
    const int b   = blockIdx.x >> 9;
    const int w0  = wq * 32;

    const int p      = tid & 31;          // pixel within block
    const int grp    = tid >> 5;          // wave id
    const int p_half = p >> 4;
    const int n      = p & 15;
    const int lane   = tid & 31;
    const int laneHi = lane >> 4;
    const float* xb  = x + ((long)(b << 6)) * HW;

    if (tid < 32) modSum[tid] = 0.f;

    // Prefetch this wave's phase-2 A fragments into cache (global_prefetch_b8)
    {
        const unsigned short* apf = wsA + (((grp & 3) * 18) * 32 + lane) * 16;
#pragma unroll
        for (int t_k = 0; t_k < 18; t_k += 4)
            __builtin_prefetch(apf + t_k * 32 * 16, 0, 3);
    }

    // ---- Phase 0a: regular-tap im2col (zero-padded) into B-fragment layout ----
    {
        const int w = w0 + p;
#pragma unroll 1
        for (int k2 = 0; k2 < 9; ++k2) {
            int dy = k2 / 3 - 1, dx = k2 % 3 - 1;
            int yy = h + dy, xx = w + dx;
            bool ok = (yy >= 0) && (yy < 128) && (xx >= 0) && (xx < 128);
            int idx = (ok ? yy * 128 + xx : 0);
#pragma unroll
            for (int cc = 0; cc < 8; ++cc) {
                int c = grp * 8 + cc;
                float val = ok ? xb[c * HW + idx] : 0.f;
                int ck = c * 9 + k2;
                int t_k = ck >> 5;
                int kl  = ck & 31;
                int vHigh = kl >> 4;
                int r  = kl & 15;
                int lHi = r >> 3;
                int rr = r & 7;
                int s  = ((vHigh * 4 + (rr >> 1)) << 1) + (rr & 1);
                int ln = lHi * 16 + n;
                Sfrag[(((p_half * 18 + t_k) * 32 + ln) << 4) + s] = f2bf(val);
            }
        }
    }
    __syncthreads();

    // ---- Phase 0b: offset/mod GEMM on the matrix pipe (waves 0..3) ----
    if (grp < 4) {
        const int o_tile = grp & 1;                    // 2 tiles cover 32 rows
        const int ph     = grp >> 1;                   // pixel half
        const v16bf* aptr = reinterpret_cast<const v16bf*>(wsB)
                            + (o_tile * 18) * 32 + lane;
        const v16bf* bptr = reinterpret_cast<const v16bf*>(Sfrag)
                            + (ph * 18) * 32 + lane;
        v8f acc = {0.f, 0.f, 0.f, 0.f, 0.f, 0.f, 0.f, 0.f};
#pragma unroll 3
        for (int t_k = 0; t_k < 18; ++t_k) {
            v16bf a  = aptr[t_k * 32];
            v16bf bb = bptr[t_k * 32];
            acc = __builtin_amdgcn_wmma_f32_16x16x32_bf16(
                false, a, false, bb, (short)0, acc, false, false);
        }
        const int pp = ph * 16 + n;
#pragma unroll
        for (int r = 0; r < 8; ++r) {
            int oc = o_tile * 16 + laneHi * 8 + r;     // C/D layout row
            float v = acc[r];
            if (oc < 9)       offYL[oc][pp]     = v + offset_b[oc];
            else if (oc < 18) offXL[oc - 9][pp] = v + offset_b[oc];
            else if (oc < 27)
                atomicAdd(&modSum[pp], fast_sigmoid(v + mod_b[oc - 18]));
        }
    }
    __syncthreads();

    // ---- Phase 1a: bilinear coords for 9 taps x 32 pixels ----
    for (int pair = tid; pair < 288; pair += 256) {
        int k2 = pair >> 5;
        int pq = pair & 31;
        int w  = w0 + pq;
        float sy = (float)h + (float)(k2 / 3 - 1) + offYL[k2][pq];
        float sx = (float)w + (float)(k2 % 3 - 1) + offXL[k2][pq];
        float y0f = floorf(sy), x0f = floorf(sx);
        float wy1 = sy - y0f, wx1 = sx - x0f;
        float wy0 = 1.f - wy1, wx0 = 1.f - wx1;
        int y0 = (int)y0f, x0 = (int)x0f;
#pragma unroll
        for (int cnr = 0; cnr < 4; ++cnr) {
            int iy = (cnr < 2) ? y0 : (y0 + 1);
            int ix = (cnr & 1) ? (x0 + 1) : x0;
            float wgt = ((cnr < 2) ? wy0 : wy1) * ((cnr & 1) ? wx1 : wx0);
            bool ok = (iy >= 0) && (iy < 128) && (ix >= 0) && (ix < 128);
            int iyc = min(max(iy, 0), 127), ixc = min(max(ix, 0), 127);
            cIdx[pair][cnr] = iyc * 128 + ixc;
            cW[pair][cnr]   = ok ? wgt : 0.f;          // fold zero-pad mask in
        }
    }
    __syncthreads();

    // ---- Phase 1b: deformed gather + lerp, rebuild Sfrag ----
    {
#pragma unroll 1
        for (int k2 = 0; k2 < 9; ++k2) {
            int pair = (k2 << 5) + p;
            int i0 = cIdx[pair][0], i1 = cIdx[pair][1];
            int i2 = cIdx[pair][2], i3 = cIdx[pair][3];
            float g0 = cW[pair][0], g1 = cW[pair][1];
            float g2 = cW[pair][2], g3 = cW[pair][3];
#pragma unroll
            for (int cc = 0; cc < 8; ++cc) {
                int c = grp * 8 + cc;
                const float* xc = xb + c * HW;
                float val = xc[i0] * g0 + xc[i1] * g1 + xc[i2] * g2 + xc[i3] * g3;
                int ck = c * 9 + k2;
                int t_k = ck >> 5;
                int kl  = ck & 31;
                int vHigh = kl >> 4;
                int r  = kl & 15;
                int lHi = r >> 3;
                int rr = r & 7;
                int s  = ((vHigh * 4 + (rr >> 1)) << 1) + (rr & 1);
                int ln = lHi * 16 + n;
                Sfrag[(((p_half * 18 + t_k) * 32 + ln) << 4) + s] = f2bf(val);
            }
        }
    }
    __syncthreads();

    // ---- Phase 2: main GEMM, 8 waves x 18 WMMA + modulated epilogue ----
    {
        const int o_tile = grp & 3;                    // 4 tiles cover O=64
        const int ph     = grp >> 2;                   // 2 halves cover 32 px
        const v16bf* aptr = reinterpret_cast<const v16bf*>(wsA)
                            + (o_tile * 18) * 32 + lane;
        const v16bf* bptr = reinterpret_cast<const v16bf*>(Sfrag)
                            + (ph * 18) * 32 + lane;
        v8f acc = {0.f, 0.f, 0.f, 0.f, 0.f, 0.f, 0.f, 0.f};
#pragma unroll 3
        for (int t_k = 0; t_k < 18; ++t_k) {
            v16bf a  = aptr[t_k * 32];
            v16bf bb = bptr[t_k * 32];
            acc = __builtin_amdgcn_wmma_f32_16x16x32_bf16(
                false, a, false, bb, (short)0, acc, false, false);
        }
        const int pp  = ph * 16 + n;
        const int w   = w0 + pp;
        const float mod = modSum[pp] * (1.f / 9.f);
#pragma unroll
        for (int r = 0; r < 8; ++r) {
            int o = o_tile * 16 + laneHi * 8 + r;      // C/D layout: M = r + 8*laneHi
            out[(((long)((b << 6) + o)) << 14) + (h << 7) + w] = acc[r] * mod + bias[o];
        }
    }
}

extern "C" void kernel_launch(void* const* d_in, const int* in_sizes, int n_in,
                              void* d_out, int out_size, void* d_ws, size_t ws_size,
                              hipStream_t stream) {
    const float* x        = (const float*)d_in[0];
    const float* weight   = (const float*)d_in[1];
    const float* bias     = (const float*)d_in[2];
    const float* offset_w = (const float*)d_in[3];
    const float* offset_b = (const float*)d_in[4];
    const float* mod_w    = (const float*)d_in[5];
    const float* mod_b    = (const float*)d_in[6];
    unsigned short* ws    = (unsigned short*)d_ws;     // wsA (36864) + wsB (18432)
    float* out            = (float*)d_out;

    const unsigned short* wsA = ws;
    const unsigned short* wsB = ws + 4 * 18 * 32 * 16;

    dcn_wprep<<<216, 256, 0, stream>>>(weight, offset_w, mod_w, ws);
    // 4 (b) * 128 (h) * 4 (w-quads of 32 px) = 2048 blocks
    dcn_main<<<2048, 256, 0, stream>>>(x, bias, offset_b, mod_b, wsA, wsB, out);
}
